// SO3TensorProductLayer_2791728742860
// MI455X (gfx1250) — matrix operand order repaired and checked by reading, not compiled, not run on hardware
//
#include <hip/hip_runtime.h>

typedef __bf16 bf16_t;
typedef __attribute__((ext_vector_type(16))) __bf16 v16bf;
typedef __attribute__((ext_vector_type(8)))  __bf16 v8bf;
typedef __attribute__((ext_vector_type(8)))  float  v8f;
typedef __attribute__((ext_vector_type(4)))  float  v4f;
typedef __attribute__((ext_vector_type(4)))  unsigned int v4u;
typedef __attribute__((ext_vector_type(8)))  unsigned int v8u;

#define BATCH 4096
#define D_IN  128
#define H1    512           // 2*OUTPUT_DIM
#define D_OUT 256
#define K1    (D_IN * D_IN) // 16384
#define SQRT_D 11.313708498984761f   // sqrt(128): a = (sqrt(d)xi)*(sqrt(d)xj) = d*xi*xj

// Packed bf16 multiply (VOP3P V_PK_MUL_BF16) on two bf16 pairs held in a dword.
__device__ __forceinline__ unsigned int pk_mul_bf16(unsigned int a, unsigned int b)
{
  unsigned int d;
  asm("v_pk_mul_bf16 %0, %1, %2" : "=v"(d) : "v"(a), "v"(b));
  return d;
}

// Async DMA: copy 16 bytes from global memory to LDS (ASYNCcnt-tracked).
__device__ __forceinline__ void async_copy_b128(unsigned int lds_byte_addr,
                                                const void* gaddr)
{
  asm volatile("global_load_async_to_lds_b128 %0, %1, off"
               :: "v"(lds_byte_addr), "v"(gaddr) : "memory");
}

__device__ __forceinline__ void wait_async0()
{
  asm volatile("s_wait_asynccnt 0x0" ::: "memory");
}

// ---------------------------------------------------------------------------
// Transpose + f32 -> bf16 convert:  src [K][N] f32  ->  dst [N][K] bf16
// ---------------------------------------------------------------------------
__global__ __launch_bounds__(256) void transpose_cvt_kernel(
    const float* __restrict__ src, bf16_t* __restrict__ dst, int K, int N)
{
  __shared__ float tile[32][33];
  const int kb = blockIdx.x * 32;
  const int nb = blockIdx.y * 32;
  const int tx = threadIdx.x;  // 0..31
  const int ty = threadIdx.y;  // 0..7
  #pragma unroll
  for (int s = 0; s < 4; ++s) {
    int k = kb + ty + s * 8;
    tile[ty + s * 8][tx] = src[(size_t)k * N + nb + tx];   // coalesced along N
  }
  __syncthreads();
  #pragma unroll
  for (int s = 0; s < 4; ++s) {
    int n = nb + ty + s * 8;
    dst[(size_t)n * K + kb + tx] = (bf16_t)tile[tx][ty + s * 8]; // coalesced along K
  }
}

// ---------------------------------------------------------------------------
// GEMM1: H[4096][512] = relu( flat(x⊗x·d) @ W1 + b1 ).
// A generated on the fly (bf16) from an LDS-resident sqrt(d)-scaled x tile.
// B (W1t) streamed global->LDS with async copies, double buffered; DMA of
// chunk i+1 overlaps the 32 WMMAs of chunk i.
// 4 waves/block; each wave owns a 32(M) x 64(N) C tile -> 8 WMMAs per K-step.
// ---------------------------------------------------------------------------
__device__ __forceinline__ void store_h_tile(const v8f& acc, int grow_base, int col,
                                             const float* __restrict__ b1,
                                             bf16_t* __restrict__ H)
{
  const float bias = b1[col];
  #pragma unroll
  for (int v = 0; v < 8; ++v) {
    float val = acc[v] + bias;
    val = val > 0.0f ? val : 0.0f;
    H[(size_t)(grow_base + v) * H1 + col] = (bf16_t)val;
  }
}

__global__ __launch_bounds__(128) void gemm1_kernel(
    const float* __restrict__ x, const bf16_t* __restrict__ W1t,
    const float* __restrict__ b1, bf16_t* __restrict__ H)
{
  __shared__ bf16_t xs[128 * 128];        // 32 KB: bf16(sqrt(d) * x) tile
  __shared__ bf16_t bbuf[2][64 * 128];    // 2 x 16 KB: B chunk [n:64][k:128]
  const int tid  = threadIdx.x;
  const int wave = tid >> 5;              // 0..3
  const int lane = tid & 31;
  const int rowBase = blockIdx.x * 128;
  const int nBase   = blockIdx.y * 64;

  // Issue async DMA of B chunk `i` (K = i*128..+127, rows nBase..+63) into bbuf[buf].
  // 16 KB / 128 threads = 8 x 16B segments per thread.
  auto issue_chunk = [&](int i, int buf) {
    #pragma unroll
    for (int r = 0; r < 8; ++r) {
      const int s   = tid + 128 * r;      // 0..1023
      const int row = s >> 4;             // 0..63
      const int seg = s & 15;             // 0..15 (8 bf16 each)
      const bf16_t* g = W1t + (size_t)(nBase + row) * K1 + i * 128 + seg * 8;
      unsigned int lds = (unsigned int)(uintptr_t)&bbuf[buf][row * 128 + seg * 8];
      async_copy_b128(lds, g);
    }
  };

  issue_chunk(0, 0);   // overlap first DMA with the x-tile conversion below

  // Load + scale + convert the 128x128 x tile (4 consecutive floats per step).
  #pragma unroll 4
  for (int e = tid * 4; e < 128 * 128; e += 128 * 4) {
    v4f v = *(const v4f*)(x + (size_t)(rowBase + (e >> 7)) * D_IN + (e & 127));
    #pragma unroll
    for (int t = 0; t < 4; ++t) xs[e + t] = (bf16_t)(v[t] * SQRT_D);
  }

  const int waveM = wave * 32;                   // 0,32,64,96
  const int l16   = lane & 15;
  const int khalf = lane >> 4;                   // 0: K-low half, 1: K-high half
  const int kh8   = khalf * 8;

  const bf16_t* xrow0 = &xs[(waveM +      l16) * 128];
  const bf16_t* xrow1 = &xs[(waveM + 16 + l16) * 128];

  v8f c0[4] = {};   // rows waveM..+15 (per C layout), cols nBase+16q
  v8f c1[4] = {};   // rows waveM+16..+31

  for (int i = 0; i < 128; ++i) {
    wait_async0();        // our chunk-i segments are LDS-resident
    __syncthreads();      // everyone's are; and nobody still reads buf[(i+1)&1]
    if (i + 1 < 128) issue_chunk(i + 1, (i + 1) & 1);   // overlaps compute below

    const bf16_t* bb = &bbuf[i & 1][0];
    // broadcast sqrt(d)*x[row,i] into both halves of a dword (bf16 pair)
    const unsigned int u0 = (unsigned int)__builtin_bit_cast(unsigned short, xrow0[i]);
    const unsigned int u1 = (unsigned int)__builtin_bit_cast(unsigned short, xrow1[i]);
    const unsigned int xi0 = u0 | (u0 << 16);
    const unsigned int xi1 = u1 | (u1 << 16);

    #pragma unroll
    for (int jb = 0; jb < 4; ++jb) {
      const int j0 = jb * 32;
      // A layout (16-bit A 16x32): elem t -> K=khalf*16+t (contiguous, matches B).
      v4u g0lo = *(const v4u*)(xrow0 + j0 + khalf * 16);
      v4u g0hi = *(const v4u*)(xrow0 + j0 + khalf * 16 + 8);
      v4u g1lo = *(const v4u*)(xrow1 + j0 + khalf * 16);
      v4u g1hi = *(const v4u*)(xrow1 + j0 + khalf * 16 + 8);
      v8u au0, au1;
      #pragma unroll
      for (int t = 0; t < 4; ++t) {
        au0[t]     = pk_mul_bf16(xi0, g0lo[t]);
        au0[4 + t] = pk_mul_bf16(xi0, g0hi[t]);
        au1[t]     = pk_mul_bf16(xi1, g1lo[t]);
        au1[4 + t] = pk_mul_bf16(xi1, g1hi[t]);
      }
      v16bf a0 = __builtin_bit_cast(v16bf, au0);
      v16bf a1 = __builtin_bit_cast(v16bf, au1);
      // B from LDS: lane = column 16q+l16, 16 contiguous K (32B aligned)
      #pragma unroll
      for (int q = 0; q < 4; ++q) {
        v16bf b = *(const v16bf*)(bb + (16 * q + l16) * 128 + j0 + khalf * 16);
        c0[q] = __builtin_amdgcn_wmma_f32_16x16x32_bf16(false, a0, false, b, (short)0, c0[q], false, false);
        c1[q] = __builtin_amdgcn_wmma_f32_16x16x32_bf16(false, a1, false, b, (short)0, c1[q], false, false);
      }
    }
  }

  // C layout: VGPR v holds M = v + khalf*8 (per 16-row subtile), N = lane%16
  const int grow0 = rowBase + waveM + khalf * 8;
  const int grow1 = rowBase + waveM + 16 + khalf * 8;
  #pragma unroll
  for (int q = 0; q < 4; ++q) {
    store_h_tile(c0[q], grow0, nBase + 16 * q + l16, b1, H);
    store_h_tile(c1[q], grow1, nBase + 16 * q + l16, b1, H);
  }
}

// ---------------------------------------------------------------------------
// GEMM2: out[4096][256] = H(bf16) @ W2 + b2.  Same 32x64 wave tiling.
// ---------------------------------------------------------------------------
__global__ __launch_bounds__(128) void gemm2_kernel(
    const bf16_t* __restrict__ Hm, const bf16_t* __restrict__ W2t,
    const float* __restrict__ b2, float* __restrict__ out)
{
  const int tid  = threadIdx.x;
  const int wave = tid >> 5;         // 0..3
  const int lane = tid & 31;
  const int rowBase = blockIdx.x * 128;
  const int nBase   = blockIdx.y * 64;
  const int waveM = wave * 32;
  const int l16   = lane & 15;
  const int khalf = lane >> 4;
  const int kh8   = khalf * 8;

  const bf16_t* hrow0 = Hm + (size_t)(rowBase + waveM +      l16) * H1;
  const bf16_t* hrow1 = Hm + (size_t)(rowBase + waveM + 16 + l16) * H1;
  const bf16_t* Bp[4];
  #pragma unroll
  for (int q = 0; q < 4; ++q)
    Bp[q] = W2t + (size_t)(nBase + 16 * q + l16) * H1 + khalf * 16;

  v8f c0[4] = {};
  v8f c1[4] = {};

  #pragma unroll 4
  for (int kb = 0; kb < H1 / 32; ++kb) {
    const int k0 = kb * 32;
    v8bf a0lo = *(const v8bf*)(hrow0 + k0 + khalf * 16);
    v8bf a0hi = *(const v8bf*)(hrow0 + k0 + khalf * 16 + 8);
    v8bf a1lo = *(const v8bf*)(hrow1 + k0 + khalf * 16);
    v8bf a1hi = *(const v8bf*)(hrow1 + k0 + khalf * 16 + 8);
    v16bf a0 = __builtin_shufflevector(a0lo, a0hi, 0,1,2,3,4,5,6,7,8,9,10,11,12,13,14,15);
    v16bf a1 = __builtin_shufflevector(a1lo, a1hi, 0,1,2,3,4,5,6,7,8,9,10,11,12,13,14,15);
    #pragma unroll
    for (int q = 0; q < 4; ++q) {
      v16bf b = *(const v16bf*)(Bp[q] + k0);
      c0[q] = __builtin_amdgcn_wmma_f32_16x16x32_bf16(false, a0, false, b, (short)0, c0[q], false, false);
      c1[q] = __builtin_amdgcn_wmma_f32_16x16x32_bf16(false, a1, false, b, (short)0, c1[q], false, false);
    }
  }

  const int grow0 = rowBase + waveM + khalf * 8;
  const int grow1 = grow0 + 16;
  #pragma unroll
  for (int q = 0; q < 4; ++q) {
    const int col = nBase + 16 * q + l16;
    const float bias = b2[col];
    #pragma unroll
    for (int v = 0; v < 8; ++v) {
      out[(size_t)(grow0 + v) * D_OUT + col] = c0[q][v] + bias;
      out[(size_t)(grow1 + v) * D_OUT + col] = c1[q][v] + bias;
    }
  }
}

// ---------------------------------------------------------------------------
extern "C" void kernel_launch(void* const* d_in, const int* in_sizes, int n_in,
                              void* d_out, int out_size, void* d_ws, size_t ws_size,
                              hipStream_t stream)
{
  const float* x  = (const float*)d_in[0];
  const float* W1 = (const float*)d_in[1];
  const float* b1 = (const float*)d_in[2];
  const float* W2 = (const float*)d_in[3];
  const float* b2 = (const float*)d_in[4];
  float* out = (float*)d_out;

  char* ws = (char*)d_ws;
  bf16_t* W1t = (bf16_t*)(ws);                                         // 16 MB
  bf16_t* W2t = (bf16_t*)(ws + (size_t)K1 * H1 * 2);                   // 256 KB (512K slot)
  bf16_t* Hm  = (bf16_t*)(ws + (size_t)K1 * H1 * 2 + 512 * 1024);      // 4 MB

  dim3 tb(32, 8);
  transpose_cvt_kernel<<<dim3(K1 / 32, H1 / 32), tb, 0, stream>>>(W1, W1t, K1, H1);
  transpose_cvt_kernel<<<dim3(H1 / 32, D_OUT / 32), tb, 0, stream>>>(W2, W2t, H1, D_OUT);
  gemm1_kernel<<<dim3(BATCH / 128, H1 / 64), 128, 0, stream>>>(x, W1t, b1, Hm);
  gemm2_kernel<<<dim3(BATCH / 128, D_OUT / 64), 128, 0, stream>>>(Hm, W2t, b2, out);
}
